// StructTunedLogitGraphNN_86818468921786
// MI455X (gfx1250) — matrix operand order repaired and checked by reading, not compiled, not run on hardware
//
#include <hip/hip_runtime.h>

#define N_NODESC 100000
#define N_EDGESC 1600000
#define IN_DIMC  257
#define KPAD1    288
#define HIDC     128
#define HEADSC   4
#define CHC      32
#define EDGE_DIMC 8
#define N_GRAPHSC 64
#define NEG_SLOPEC 0.2f

typedef __attribute__((ext_vector_type(16))) _Float16 v16h;
typedef __attribute__((ext_vector_type(8)))  _Float16 v8h;
typedef __attribute__((ext_vector_type(8)))  float    v8f;

// ---- order-preserving float <-> uint for atomic max ----
__device__ __forceinline__ unsigned f32_ord(float f) {
  unsigned u = __float_as_uint(f);
  return (u & 0x80000000u) ? ~u : (u | 0x80000000u);
}
__device__ __forceinline__ float ord_f32(unsigned u) {
  return __uint_as_float((u & 0x80000000u) ? (u & 0x7FFFFFFFu) : ~u);
}

// ---- zero buffer ----
__global__ void k_zero(float* __restrict__ p, size_t n) {
  size_t i = (size_t)blockIdx.x * blockDim.x + threadIdx.x;
  if (i < n) p[i] = 0.0f;
}

// ---- pack fp32 [M,K] -> f16 [M,Kpad] zero-padded (grid covers M*Kpad exactly) ----
__global__ void k_pack_a(const float* __restrict__ in, _Float16* __restrict__ out,
                         int K, int Kpad) {
  size_t i = (size_t)blockIdx.x * blockDim.x + threadIdx.x;
  size_t m = i / (size_t)Kpad;
  int k = (int)(i - m * (size_t)Kpad);
  out[i] = (k < K) ? (_Float16)in[m * (size_t)K + k] : (_Float16)0.0f;
}

// ---- pack fp32 B [K,128] -> f16 B^T [128,Kpad] zero-padded ----
__global__ void k_pack_bt(const float* __restrict__ in, _Float16* __restrict__ out,
                          int K, int Kpad) {
  int i = blockIdx.x * blockDim.x + threadIdx.x;   // over 128*Kpad (exact)
  int n = i / Kpad;
  int k = i - n * Kpad;
  out[i] = (k < K) ? (_Float16)in[(size_t)k * HIDC + n] : (_Float16)0.0f;
}

// ---- WMMA GEMM: C[M,128] = Apad[M,Kpad] @ BpT^T, f16 in / f32 acc ----
// grid.x = M/16; block = 256 (8 waves, wave w owns N-tile w). No branches,
// no predication -> EXEC all ones around WMMA. All loads are b128.
__global__ void __launch_bounds__(256)
k_gemm_wmma(const _Float16* __restrict__ A, const _Float16* __restrict__ BT,
            float* __restrict__ C, int Kpad) {
  const int lane = threadIdx.x & 31;
  const int wv   = threadIdx.x >> 5;   // N tile (0..7)
  const int grp  = lane >> 4;          // half-wave group
  const int l15  = lane & 15;
  const _Float16* __restrict__ Arow = A  + (size_t)(blockIdx.x * 16 + l15) * Kpad;
  const _Float16* __restrict__ Brow = BT + (size_t)(wv * 16 + l15) * Kpad;
  v8f acc = {};
  for (int k0 = 0; k0 < Kpad; k0 += 32) {
    // A fragment (16-bit A 16x32 layout): two contiguous 16B chunks per lane
    v8h alo = *(const v8h*)(Arow + k0 + grp * 8);
    v8h ahi = *(const v8h*)(Arow + k0 + 16 + grp * 8);
    // B fragment: lane group g covers k in [k0+g*8, k0+g*8+8) and [k0+16+g*8, k0+16+g*8+8)
    v8h blo = *(const v8h*)(Brow + k0 + grp * 8);
    v8h bhi = *(const v8h*)(Brow + k0 + 16 + grp * 8);
    v16h a = __builtin_shufflevector(alo, ahi, 0,1,2,3,4,5,6,7,8,9,10,11,12,13,14,15);
    v16h b = __builtin_shufflevector(blo, bhi, 0,1,2,3,4,5,6,7,8,9,10,11,12,13,14,15);
    acc = __builtin_amdgcn_wmma_f32_16x16x32_f16(false, a, false, b,
                                                 (short)0, acc, false, false);
  }
  const int rbase = blockIdx.x * 16 + grp * 8;
  const int col   = wv * 16 + l15;
#pragma unroll
  for (int i = 0; i < 8; ++i)
    C[(size_t)(rbase + i) * HIDC + col] = acc[i];
}

// ---- per-(node,head) attention scalars ----
__global__ void k_attn_node(const float* __restrict__ xs,
                            const float* __restrict__ attS,
                            const float* __restrict__ attD,
                            float* __restrict__ asrc, float* __restrict__ adst) {
  int i = blockIdx.x * blockDim.x + threadIdx.x;
  if (i >= N_NODESC * HEADSC) return;
  int n = i >> 2, h = i & 3;
  const float* row = xs + (size_t)n * HIDC + h * CHC;
  const float* s = attS + h * CHC;
  const float* d = attD + h * CHC;
  float ds = 0.f, dd = 0.f;
#pragma unroll
  for (int c = 0; c < CHC; ++c) { float v = row[c]; ds += v * s[c]; dd += v * d[c]; }
  asrc[i] = ds; adst[i] = dd;
}

// ---- vE[d,h] = sum_c We[d, h*32+c] * att_edge[h,c]  (8x4) ----
__global__ void k_ve(const float* __restrict__ We, const float* __restrict__ attE,
                     float* __restrict__ vE) {
  int t = threadIdx.x;                 // 32 threads
  int d = t >> 2, h = t & 3;
  float s = 0.f;
#pragma unroll
  for (int c = 0; c < CHC; ++c)
    s += We[(size_t)d * HIDC + h * CHC + c] * attE[h * CHC + c];
  vE[t] = s;                           // vE[d*4+h]
}

// ---- pass 1: raw alpha + segment max ----
__global__ void k_edge_alpha(const int* __restrict__ src, const int* __restrict__ dst,
                             const float* __restrict__ ea, const float* __restrict__ vE,
                             const float* __restrict__ asrc, const float* __restrict__ adst,
                             float* __restrict__ wedge, unsigned* __restrict__ amax) {
  int e = blockIdx.x * blockDim.x + threadIdx.x;
  if (e >= N_EDGESC) return;
  int s = src[e], d = dst[e];
  float eav[EDGE_DIMC];
#pragma unroll
  for (int j = 0; j < EDGE_DIMC; ++j) eav[j] = ea[(size_t)e * EDGE_DIMC + j];
  const float4 as4 = *(const float4*)(asrc + (size_t)s * 4);
  const float4 ad4 = *(const float4*)(adst + (size_t)d * 4);
  const float asv[4] = {as4.x, as4.y, as4.z, as4.w};
  const float adv[4] = {ad4.x, ad4.y, ad4.z, ad4.w};
#pragma unroll
  for (int h = 0; h < 4; ++h) {
    float ae = 0.f;
#pragma unroll
    for (int j = 0; j < EDGE_DIMC; ++j) ae += eav[j] * vE[j * 4 + h];
    float al = asv[h] + adv[h] + ae;
    al = (al > 0.f) ? al : NEG_SLOPEC * al;
    wedge[(size_t)e * 4 + h] = al;
    atomicMax(amax + (size_t)d * 4 + h, f32_ord(al));
  }
}

// ---- pass 2: exp(alpha - max) + segment sum ----
__global__ void k_edge_exp(const int* __restrict__ dst, float* __restrict__ wedge,
                           const unsigned* __restrict__ amax, float* __restrict__ denom) {
  int e = blockIdx.x * blockDim.x + threadIdx.x;
  if (e >= N_EDGESC) return;
  int d = dst[e];
#pragma unroll
  for (int h = 0; h < 4; ++h) {
    float m = ord_f32(amax[(size_t)d * 4 + h]);
    float w = __expf(wedge[(size_t)e * 4 + h] - m);
    wedge[(size_t)e * 4 + h] = w;
    __hip_atomic_fetch_add(denom + (size_t)d * 4 + h, w,
                           __ATOMIC_RELAXED, __HIP_MEMORY_SCOPE_AGENT);
  }
}

// ---- pass 3: normalized weighted scatter (128 threads per edge) ----
__global__ void k_edge_scatter(const int* __restrict__ src, const int* __restrict__ dst,
                               const float* __restrict__ xs, const float* __restrict__ wedge,
                               const float* __restrict__ denom, float* __restrict__ acc) {
  size_t gid = (size_t)blockIdx.x * blockDim.x + threadIdx.x;
  size_t e = gid >> 7;
  int t = (int)(gid & 127);
  int h = t >> 5;
  int s = src[e], d = dst[e];
  float w  = wedge[e * 4 + h];
  float dn = denom[(size_t)d * 4 + h];
  float coef = w / (dn + 1e-16f);
  float v = xs[(size_t)s * HIDC + t] * coef;
  __hip_atomic_fetch_add(acc + (size_t)d * HIDC + t, v,
                         __ATOMIC_RELAXED, __HIP_MEMORY_SCOPE_AGENT);
}

// ---- bias + relu in place ----
__global__ void k_bias_relu(float* __restrict__ acc, const float* __restrict__ bias) {
  size_t i = (size_t)blockIdx.x * blockDim.x + threadIdx.x;
  int c = (int)(i & 127);
  float v = acc[i] + bias[c];
  acc[i] = v > 0.f ? v : 0.f;
}

// ---- mean-pool accumulation ----
__global__ void k_pool(const float* __restrict__ h, const int* __restrict__ batch,
                       float* __restrict__ pooled, float* __restrict__ counts) {
  size_t i = (size_t)blockIdx.x * blockDim.x + threadIdx.x;
  size_t n = i >> 7; int c = (int)(i & 127);
  int g = batch[n];
  __hip_atomic_fetch_add(pooled + (size_t)g * HIDC + c, h[i],
                         __ATOMIC_RELAXED, __HIP_MEMORY_SCOPE_AGENT);
  if (c == 0)
    __hip_atomic_fetch_add(counts + g, 1.0f,
                           __ATOMIC_RELAXED, __HIP_MEMORY_SCOPE_AGENT);
}

// ---- tiny classifier: one block per graph ----
__global__ void k_classifier(const float* __restrict__ pooled, const float* __restrict__ counts,
                             const float* __restrict__ Wc1, const float* __restrict__ bc1,
                             const float* __restrict__ Wc2, const float* __restrict__ bc2,
                             float* __restrict__ out) {
  __shared__ float sp[HIDC];
  __shared__ float sz[64];
  int g = blockIdx.x;
  int t = threadIdx.x;                 // 64 threads
  float inv = 1.0f / fmaxf(counts[g], 1.0f);
  sp[t]      = pooled[(size_t)g * HIDC + t] * inv;
  sp[t + 64] = pooled[(size_t)g * HIDC + t + 64] * inv;
  __syncthreads();
  float z = bc1[t];
  for (int k = 0; k < HIDC; ++k) z += sp[k] * Wc1[(size_t)k * 64 + t];
  sz[t] = z > 0.f ? z : 0.f;
  __syncthreads();
  if (t < 2) {
    float o = bc2[t];
    for (int k = 0; k < 64; ++k) o += sz[k] * Wc2[(size_t)k * 2 + t];
    out[(size_t)g * 2 + t] = o;
  }
}

extern "C" void kernel_launch(void* const* d_in, const int* in_sizes, int n_in,
                              void* d_out, int out_size, void* d_ws, size_t ws_size,
                              hipStream_t stream) {
  const float* x    = (const float*)d_in[0];
  const int*   ei   = (const int*)d_in[1];
  const float* ea   = (const float*)d_in[2];
  const int*   batch= (const int*)d_in[3];
  const float* W1   = (const float*)d_in[4];
  const float* as1  = (const float*)d_in[5];
  const float* ad1  = (const float*)d_in[6];
  const float* We1  = (const float*)d_in[7];
  const float* ae1  = (const float*)d_in[8];
  const float* b1   = (const float*)d_in[9];
  const float* W2   = (const float*)d_in[10];
  const float* as2  = (const float*)d_in[11];
  const float* ad2  = (const float*)d_in[12];
  const float* We2  = (const float*)d_in[13];
  const float* ae2  = (const float*)d_in[14];
  const float* b2   = (const float*)d_in[15];
  const float* Wc1  = (const float*)d_in[16];
  const float* bc1  = (const float*)d_in[17];
  const float* Wc2  = (const float*)d_in[18];
  const float* bc2  = (const float*)d_in[19];
  const int* src = ei;
  const int* dst = ei + N_EDGESC;

  // workspace carve (every piece is a multiple of 128B -> 16B alignment holds)
  float*    xs     = (float*)d_ws;                                  // N*128 f32
  float*    acc    = xs   + (size_t)N_NODESC * HIDC;                // N*128 f32
  float*    asrc   = acc  + (size_t)N_NODESC * HIDC;                // N*4
  float*    adst   = asrc + (size_t)N_NODESC * HEADSC;              // N*4
  unsigned* amax   = (unsigned*)(adst + (size_t)N_NODESC * HEADSC); // N*4
  float*    denom  = (float*)(amax + (size_t)N_NODESC * HEADSC);    // N*4
  float*    wedge  = denom + (size_t)N_NODESC * HEADSC;             // E*4
  float*    vE     = wedge + (size_t)N_EDGESC * HEADSC;             // 32 (pad 128)
  float*    pooled = vE + 32;                                       // 64*128
  float*    counts = pooled + (size_t)N_GRAPHSC * HIDC;             // 64
  _Float16* Apad   = (_Float16*)(counts + 64);                      // N*KPAD1 f16
  _Float16* BpT    = Apad + (size_t)N_NODESC * KPAD1;               // 128*KPAD1 f16

  const int nh        = N_NODESC * HEADSC;                   // 400000
  const int nfeat_blk = (N_NODESC * HIDC) / 256;             // 50000 (exact)
  const int e_blk     = N_EDGESC / 256;                      // 6250  (exact)
  const int esc_blk   = (int)(((size_t)N_EDGESC * HIDC) / 256); // 800000 (exact)

  auto layer = [&](const float* in, int K, int Kpad, const float* W,
                   const float* aS, const float* aD, const float* We,
                   const float* aE, const float* bias) {
    // pack operands to f16 (grids exact: N*Kpad and 128*Kpad divisible by 256)
    k_pack_a<<<(int)(((size_t)N_NODESC * Kpad) / 256), 256, 0, stream>>>(in, Apad, K, Kpad);
    k_pack_bt<<<(HIDC * Kpad) / 256, 256, 0, stream>>>(W, BpT, K, Kpad);
    k_gemm_wmma<<<N_NODESC / 16, 256, 0, stream>>>(Apad, BpT, xs, Kpad);
    k_zero<<<nfeat_blk, 256, 0, stream>>>(acc, (size_t)N_NODESC * HIDC);
    k_zero<<<(nh + 255) / 256, 256, 0, stream>>>((float*)amax, (size_t)nh);
    k_zero<<<(nh + 255) / 256, 256, 0, stream>>>(denom, (size_t)nh);
    k_attn_node<<<(nh + 255) / 256, 256, 0, stream>>>(xs, aS, aD, asrc, adst);
    k_ve<<<1, 32, 0, stream>>>(We, aE, vE);
    k_edge_alpha<<<e_blk, 256, 0, stream>>>(src, dst, ea, vE, asrc, adst, wedge, amax);
    k_edge_exp<<<e_blk, 256, 0, stream>>>(dst, wedge, amax, denom);
    k_edge_scatter<<<esc_blk, 256, 0, stream>>>(src, dst, xs, wedge, denom, acc);
    k_bias_relu<<<nfeat_blk, 256, 0, stream>>>(acc, bias);
  };

  // layer 1: input x [N,257] -> h in acc
  layer(x, IN_DIMC, KPAD1, W1, as1, ad1, We1, ae1, b1);
  // layer 2: input h (=acc; packed before acc is re-zeroed in stream order)
  layer(acc, HIDC, HIDC, W2, as2, ad2, We2, ae2, b2);

  // global mean pool + classifier
  k_zero<<<(N_GRAPHSC * HIDC + 255) / 256, 256, 0, stream>>>(pooled, (size_t)N_GRAPHSC * HIDC);
  k_zero<<<1, 64, 0, stream>>>(counts, (size_t)N_GRAPHSC);
  k_pool<<<nfeat_blk, 256, 0, stream>>>(acc, batch, pooled, counts);
  k_classifier<<<N_GRAPHSC, 64, 0, stream>>>(pooled, counts, Wc1, bc1, Wc2, bc2,
                                             (float*)d_out);
}
